// SpeechAttDecVol5_58884001628654
// MI455X (gfx1250) — matrix operand order repaired and checked by reading, not compiled, not run on hardware
//
#include <hip/hip_runtime.h>
#include <hip/hip_bf16.h>

// ---------------------------------------------------------------------------
// SpeechAttDecVol5 on MI455X (gfx1250, wave32, WMMA)
//
// Reference collapses: attention weights are exactly 1/L, so ctx is the mean
// of the encoder output over L and is constant across time. The recurrence is
// elementwise in the batch dim -> one persistent kernel, each block owns 16
// batch rows and runs all 38 GRU steps with v_wmma_f32_16x16x32_f16.
// Stage 2 uses paired column tiles (two independent accumulator chains).
// NOTE: k_recurrence pointers are deliberately NOT __restrict__ so the
// compiler cannot hoist the (loop-invariant) weight fragments into ~128
// persistent VGPRs across the 38-step loop, which pushed allocation past 256
// and injected s_set_vgpr_msb toggles into the WMMA chain.
// ---------------------------------------------------------------------------

typedef _Float16 f16_t;
typedef __attribute__((ext_vector_type(8)))  _Float16 v8h_t;
typedef __attribute__((ext_vector_type(16))) _Float16 v16h_t;
typedef __attribute__((ext_vector_type(8)))  float    v8f_t;

union AF { v16h_t v; v8h_t h[2]; };

static __device__ __forceinline__ v8f_t wmma16(v16h_t a, v16h_t b, v8f_t c) {
  return __builtin_amdgcn_wmma_f32_16x16x32_f16(false, a, false, b, (short)0, c,
                                                false, false);
}

#define L_ENC 39
#define NB    4096
#define H     256
#define H3    768
#define TSTEPS 38
#define NCHAR 27
#define OUTQ  10

// LDS strides (elements)
#define AS   264     // f16 row stride for h16/x16 (multiple of 8, bank-spread)
#define GSTR 1540    // fp32 row stride for gate buffer (half-waves disjoint banks)

// ---------------------------------------------------------------------------
// K1: convert weights to f16 row-major copies
// ---------------------------------------------------------------------------
__global__ __launch_bounds__(256) void k_cvt_weights(
    const float* __restrict__ W_cat, const float* __restrict__ W_ih,
    const float* __restrict__ W_hh, f16_t* __restrict__ Wch,
    f16_t* __restrict__ Wih, f16_t* __restrict__ Whh) {
  int id = blockIdx.x * 256 + threadIdx.x;
  const int n1 = H * H;          // 65536
  const int n2 = n1 + H3 * H;    // 262144
  const int n3 = n2 + H3 * H;    // 458752
  if (id < n1) {
    int r = id >> 8, c = id & 255;
    Wch[id] = (f16_t)W_cat[r * (2 * H) + c];
  } else if (id < n2) {
    int k = id - n1;
    Wih[k] = (f16_t)W_ih[k];
  } else if (id < n3) {
    int k = id - n2;
    Whh[k] = (f16_t)W_hh[k];
  }
}

// ---------------------------------------------------------------------------
// K2: ctx[n][k] = (1/L) * sum_l enc[l][n][k]
// ---------------------------------------------------------------------------
__global__ __launch_bounds__(256) void k_ctx(const float* __restrict__ enc,
                                             float* __restrict__ ctx) {
  int id = blockIdx.x * 256 + threadIdx.x;  // id < NB*H
  const size_t plane = (size_t)NB * H;
  float acc = 0.f;
#pragma unroll 1
  for (int l = 0; l < L_ENC; ++l) acc += enc[(size_t)l * plane + id];
  ctx[id] = acc * (1.0f / (float)L_ENC);
}

// ---------------------------------------------------------------------------
// K3: xc[n][j] = b_cat[j] + sum_k ctx[n][k] * W_cat[j][H+k]
// ---------------------------------------------------------------------------
__global__ __launch_bounds__(256) void k_xc(const float* __restrict__ ctx,
                                            const float* __restrict__ W_cat,
                                            const float* __restrict__ b_cat,
                                            float* __restrict__ xc) {
  __shared__ float cx[H];
  const int n = blockIdx.x;
  const int j = threadIdx.x;
  cx[j] = ctx[(size_t)n * H + j];
  __syncthreads();
  float acc = b_cat[j];
  const float* wr = W_cat + (size_t)j * (2 * H) + H;
#pragma unroll 4
  for (int k = 0; k < H; ++k) acc += cx[k] * wr[k];
  xc[(size_t)n * H + j] = acc;
}

// ---------------------------------------------------------------------------
// K4: persistent GRU recurrence. Block owns 16 batch rows, 256 threads
//     (8 waves). Per step:
//       stage1: x = h @ Wch^T + xc            (16x256, WMMA)
//       stage2: gi = x @ Wih^T + b_ih (waves 0-3)
//               gh = h @ Whh^T + b_hh (waves 4-7)   (16x768 each, WMMA)
//       stage3: gates -> h_new -> LDS + outs[t]
// ---------------------------------------------------------------------------
__global__ __launch_bounds__(256) void k_recurrence(
    const float* hidden, const float* xc,
    const f16_t* Wch, const f16_t* Wih, const f16_t* Whh,
    const float* b_ih, const float* b_hh, float* outs) {
  __shared__ f16_t sh16[16 * AS];    // h in f16 (A matrix)
  __shared__ f16_t sx16[16 * AS];    // x in f16 (A matrix)
  __shared__ float shf[16 * H];      // h in fp32
  __shared__ float gl[16 * GSTR];    // gates fp32: [0,768)=gi, [768,1536)=gh

  const int tid  = threadIdx.x;
  const int wave = tid >> 5;
  const int lane = tid & 31;
  const int lo   = lane & 15;
  const int hi   = lane >> 4;
  const int sel  = hi ? 8 : 0;       // unused; A loads offset K by hi * 16 per half-wave
  const int hi8  = hi ? 8 : 0;       // C/D row offset per half-wave
  const int n0   = blockIdx.x * 16;

  // stage-2 role, fixed per wave
  const bool isGi = (wave < 4);
  const int  ww   = isGi ? wave : (wave - 4);
  const f16_t* As2 = isGi ? sx16 : sh16;
  const f16_t* W2  = isGi ? Wih : Whh;
  const float* bias2 = isGi ? b_ih : b_hh;
  const int gbase = isGi ? 0 : H3;

  // ---- init h from input hidden (1,N,H)
  {
    const int col = tid;  // 0..255
#pragma unroll
    for (int row = 0; row < 16; ++row) {
      float v = hidden[(size_t)(n0 + row) * H + col];
      shf[row * H + col]  = v;
      sh16[row * AS + col] = (f16_t)v;
    }
  }

#pragma unroll 1
  for (int t = 0; t < TSTEPS; ++t) {
    __syncthreads();  // h (f16+fp32) ready

    // ================= stage 1: x = h @ Wch^T + xc =================
    {
      const int cb0 = wave * 32;        // two 16-col tiles per wave
      const int cb1 = cb0 + 16;
      v8f_t acc0, acc1;
#pragma unroll
      for (int v = 0; v < 8; ++v) {
        int m = v + hi8;
        acc0[v] = xc[(size_t)(n0 + m) * H + cb0 + lo];
        acc1[v] = xc[(size_t)(n0 + m) * H + cb1 + lo];
      }
#pragma unroll
      for (int kt = 0; kt < 8; ++kt) {
        AF a;
        a.h[0] = *(const v8h_t*)&sh16[lo * AS + kt * 32 + hi * 16];
        a.h[1] = *(const v8h_t*)&sh16[lo * AS + kt * 32 + hi * 16 + 8];
        v16h_t b0 = *(const v16h_t*)(Wch + (size_t)(cb0 + lo) * H + kt * 32 + hi * 16);
        v16h_t b1 = *(const v16h_t*)(Wch + (size_t)(cb1 + lo) * H + kt * 32 + hi * 16);
        acc0 = wmma16(a.v, b0, acc0);
        acc1 = wmma16(a.v, b1, acc1);
      }
#pragma unroll
      for (int v = 0; v < 8; ++v) {
        int m = v + hi8;
        sx16[m * AS + cb0 + lo] = (f16_t)acc0[v];
        sx16[m * AS + cb1 + lo] = (f16_t)acc1[v];
      }
    }
    __syncthreads();  // x ready

    // ======== stage 2: gi = x@Wih^T + b_ih  /  gh = h@Whh^T + b_hh ========
    // 12 column tiles per wave, processed in pairs -> 2 independent
    // accumulator chains to overlap WMMA latency.
    {
#pragma unroll 1
      for (int i = 0; i < 12; i += 2) {
        const int col0 = (ww * 12 + i) * 16 + lo;
        const int col1 = col0 + 16;
        const float bv0 = bias2[col0];
        const float bv1 = bias2[col1];
        v8f_t acc0, acc1;
#pragma unroll
        for (int v = 0; v < 8; ++v) { acc0[v] = bv0; acc1[v] = bv1; }
#pragma unroll
        for (int kt = 0; kt < 8; ++kt) {
          AF a;
          a.h[0] = *(const v8h_t*)&As2[lo * AS + kt * 32 + hi * 16];
          a.h[1] = *(const v8h_t*)&As2[lo * AS + kt * 32 + hi * 16 + 8];
          v16h_t b0 = *(const v16h_t*)(W2 + (size_t)col0 * H + kt * 32 + hi * 16);
          v16h_t b1 = *(const v16h_t*)(W2 + (size_t)col1 * H + kt * 32 + hi * 16);
          acc0 = wmma16(a.v, b0, acc0);
          acc1 = wmma16(a.v, b1, acc1);
        }
#pragma unroll
        for (int v = 0; v < 8; ++v) {
          gl[(v + hi8) * GSTR + gbase + col0] = acc0[v];
          gl[(v + hi8) * GSTR + gbase + col1] = acc1[v];
        }
      }
    }
    __syncthreads();  // gates ready

    // ================= stage 3: gates -> h_new =================
    {
      const int col = tid;  // 0..255
#pragma unroll
      for (int row = 0; row < 16; ++row) {
        const float* g = gl + row * GSTR;
        float gir = g[col];
        float giz = g[256 + col];
        float gin = g[512 + col];
        float ghr = g[768 + col];
        float ghz = g[1024 + col];
        float ghn = g[1280 + col];
        float hp  = shf[row * H + col];
        float r = 1.0f / (1.0f + __expf(-(gir + ghr)));
        float z = 1.0f / (1.0f + __expf(-(giz + ghz)));
        float ni = gin + r * ghn;
        float nn = 1.0f - 2.0f / (__expf(2.0f * ni) + 1.0f);  // tanh
        float hn = (1.0f - z) * nn + z * hp;
        shf[row * H + col]   = hn;
        sh16[row * AS + col] = (f16_t)hn;
        outs[((size_t)t * NB + n0 + row) * H + col] = hn;
      }
    }
  }
}

// ---------------------------------------------------------------------------
// K5: output head, one block per batch row n.
//   logits[t][c] = outs[t][n][:] . W_out[c][:] + b_out[c]       (38 x 27)
//   xq[c][o]     = sum_t logits[t][c] * W_q[o][t] + b_q[o]      (27 x 10)
//   out[n][o][c] = log_softmax_c(xq[:,o])
// ---------------------------------------------------------------------------
__global__ __launch_bounds__(256) void k_head(
    const float* __restrict__ outs, const float* __restrict__ W_out,
    const float* __restrict__ b_out, const float* __restrict__ W_q,
    const float* __restrict__ b_q, float* __restrict__ out) {
  __shared__ float ho[TSTEPS * H];       // 38x256
  __shared__ float wo[NCHAR * 257];      // padded W_out
  __shared__ float lg[TSTEPS * 28];      // padded logits
  __shared__ float xq[OUTQ * 28];        // [o][c] padded
  __shared__ float lse[OUTQ];

  const int n = blockIdx.x;
  const int tid = threadIdx.x;

  for (int idx = tid; idx < TSTEPS * H; idx += 256) {
    int t = idx >> 8, k = idx & 255;
    ho[idx] = outs[((size_t)t * NB + n) * H + k];
  }
  for (int idx = tid; idx < NCHAR * H; idx += 256) {
    int c = idx >> 8, k = idx & 255;
    wo[c * 257 + k] = W_out[(size_t)c * H + k];
  }
  __syncthreads();

  for (int idx = tid; idx < TSTEPS * NCHAR; idx += 256) {
    int t = idx / NCHAR, c = idx - t * NCHAR;
    float acc = b_out[c];
    const float* hr = ho + t * H;
    const float* wr = wo + c * 257;
#pragma unroll 4
    for (int k = 0; k < H; ++k) acc += hr[k] * wr[k];
    lg[t * 28 + c] = acc;
  }
  __syncthreads();

  if (tid < NCHAR * OUTQ) {
    int c = tid / OUTQ, o = tid - c * OUTQ;
    float acc = b_q[o];
#pragma unroll
    for (int t = 0; t < TSTEPS; ++t) acc += lg[t * 28 + c] * W_q[o * TSTEPS + t];
    xq[o * 28 + c] = acc;
  }
  __syncthreads();

  if (tid < OUTQ) {
    int o = tid;
    float m = -3.402823466e+38f;
    for (int c = 0; c < NCHAR; ++c) m = fmaxf(m, xq[o * 28 + c]);
    float s = 0.f;
    for (int c = 0; c < NCHAR; ++c) s += expf(xq[o * 28 + c] - m);
    lse[o] = m + logf(s);
  }
  __syncthreads();

  if (tid < NCHAR * OUTQ) {
    int c = tid / OUTQ, o = tid - c * OUTQ;
    out[((size_t)n * OUTQ + o) * NCHAR + c] = xq[o * 28 + c] - lse[o];
  }
}

// ---------------------------------------------------------------------------
// Launcher
// inputs: 0 enc(L,N,H) 1 hidden(1,N,H) 2 W_cat 3 b_cat 4 W_ih 5 W_hh
//         6 b_ih 7 b_hh 8 W_out 9 b_out 10 W_q 11 b_q   (all fp32)
// output: (N,10,27) fp32 log-softmax
// ---------------------------------------------------------------------------
extern "C" void kernel_launch(void* const* d_in, const int* in_sizes, int n_in,
                              void* d_out, int out_size, void* d_ws, size_t ws_size,
                              hipStream_t stream) {
  const float* enc    = (const float*)d_in[0];
  const float* hidden = (const float*)d_in[1];
  const float* W_cat  = (const float*)d_in[2];
  const float* b_cat  = (const float*)d_in[3];
  const float* W_ih   = (const float*)d_in[4];
  const float* W_hh   = (const float*)d_in[5];
  const float* b_ih   = (const float*)d_in[6];
  const float* b_hh   = (const float*)d_in[7];
  const float* W_out  = (const float*)d_in[8];
  const float* b_out  = (const float*)d_in[9];
  const float* W_q    = (const float*)d_in[10];
  const float* b_q    = (const float*)d_in[11];
  float* out = (float*)d_out;

  // workspace layout (256B aligned)
  char* ws = (char*)d_ws;
  f16_t* Wch16 = (f16_t*)(ws + 0);                 // 256*256*2   = 131072
  f16_t* Wih16 = (f16_t*)(ws + 131072);            // 768*256*2   = 393216
  f16_t* Whh16 = (f16_t*)(ws + 524288);            // 768*256*2   = 393216
  float* ctx   = (float*)(ws + 917504);            // 4096*256*4  = 4194304
  float* xcb   = (float*)(ws + 5111808);           // 4096*256*4  = 4194304
  float* outs  = (float*)(ws + 9306112);           // 38*4096*256*4 = 159383552

  k_cvt_weights<<<1792, 256, 0, stream>>>(W_cat, W_ih, W_hh, Wch16, Wih16, Whh16);
  k_ctx<<<(NB * H) / 256, 256, 0, stream>>>(enc, ctx);
  k_xc<<<NB, 256, 0, stream>>>(ctx, W_cat, b_cat, xcb);
  k_recurrence<<<NB / 16, 256, 0, stream>>>(hidden, xcb, Wch16, Wih16, Whh16,
                                            b_ih, b_hh, outs);
  k_head<<<NB, 256, 0, stream>>>(outs, W_out, b_out, W_q, b_q, out);
}
